// ViT_88416196755597
// MI455X (gfx1250) — matrix operand order repaired and checked by reading, not compiled, not run on hardware
//
#include <hip/hip_runtime.h>
#include <hip/hip_bf16.h>
#include <math.h>

typedef __attribute__((ext_vector_type(16))) _Float16 v16h;
typedef __attribute__((ext_vector_type(8)))  _Float16 v8h;
typedef __attribute__((ext_vector_type(8)))  float    v8f;
typedef __attribute__((ext_vector_type(4))) unsigned int u32x4;
typedef __attribute__((ext_vector_type(8)))  int i32x8;
typedef __attribute__((ext_vector_type(4)))  int i32x4;

#define DIMD 768
#define NHEAD 12
#define DHEAD 64
#define NTOK 199
#define BTN 40
#define TROWS (BTN * NTOK)   /* 7960 */
#define MLPD 3072

enum { AMODE_F16 = 0, AMODE_IMG = 1 };
enum { EP_F16 = 0, EP_GELU = 1, EP_RES = 2, EP_EMBED = 3 };

__device__ __forceinline__ v16h make_v16(v8h lo, v8h hi8) {
  v16h r;
#pragma unroll
  for (int j = 0; j < 8; ++j) { r[j] = lo[j]; r[j + 8] = hi8[j]; }
  return r;
}

// ---------------------------------------------------------------------------
// TDM: async 2D f16 tile load (tile_d0 x tile_d1 elements) into LDS, with
// LDS padding of 4 dwords every 16 dwords -> row pitch 40 halfs for a
// 32-half tile row (bank-conflict-free b128 LDS reads afterwards).
// Descriptor per CDNA5 ISA (D# groups 0/1; 2D -> rest 0).
// This toolchain exposes the 6-arg builtin:
//   (u32x4 g0, i32x8 g1, i32x4 g2, i32x4 g3, i32x8 extra, i32 cpol)
// ---------------------------------------------------------------------------
__device__ __forceinline__ void tdm_load_2d_f16(
    unsigned lds_addr, const _Float16* gaddr,
    unsigned tensor_d0, unsigned tensor_d1, unsigned stride_d0,
    unsigned tile_d0, unsigned tile_d1)
{
  unsigned long long ga = (unsigned long long)(size_t)gaddr;
  u32x4 g0;
  g0[0] = 1u;                                            // count=1 (valid user D#)
  g0[1] = lds_addr;                                      // lds_addr
  g0[2] = (unsigned)(ga & 0xffffffffu);                  // global_addr lo
  g0[3] = (unsigned)((ga >> 32) & 0x1ffffffu) | (2u << 30); // addr hi | type=2
  i32x8 g1;
  g1[0] = (int)((1u << 16)        // data_size = 2 bytes
              | (1u << 20)        // pad_enable
              | (3u << 22)        // pad_interval: 16 dwords
              | (3u << 25));      // pad_amount:   4 dwords
  g1[1] = (int)((tensor_d0 & 0xffffu) << 16);
  g1[2] = (int)(((tensor_d0 >> 16) & 0xffffu) | ((tensor_d1 & 0xffffu) << 16));
  g1[3] = (int)(((tensor_d1 >> 16) & 0xffffu) | ((tile_d0 & 0xffffu) << 16));
  g1[4] = (int)(tile_d1 & 0xffffu);                      // tile_dim1 (tile_dim2=0)
  g1[5] = (int)stride_d0;                                // tensor_dim0_stride lo
  g1[6] = 0;
  g1[7] = 0;
  i32x4 z4; z4[0] = 0; z4[1] = 0; z4[2] = 0; z4[3] = 0;
  i32x8 z8;
#pragma unroll
  for (int j = 0; j < 8; ++j) z8[j] = 0;
  __builtin_amdgcn_tensor_load_to_lds(g0, g1, z4, z4, z8, 0);
}

// ---------------------------------------------------------------------------
// Tiled WMMA GEMM: C(MxN) = A(MxK) * B(KxN) (+ epilogue)
// A: f16 row-major (TDM-staged) or im2col'd fp32 image (manual staging)
// B: f16 *N-major* (B16t[n*K + k], pre-converted) -> TDM-staged
// Block: 256 threads (8 waves), tile 128x128, K-step 32.
// ---------------------------------------------------------------------------
template<int AMODE, int EMODE>
__global__ __launch_bounds__(256)
void gemm_wmma(const _Float16* __restrict__ A16, const float* __restrict__ Aimg,
               const _Float16* __restrict__ B16t, const float* __restrict__ bias,
               float* __restrict__ Cf, _Float16* __restrict__ Ch,
               const float* __restrict__ pos, const float* __restrict__ tem,
               int M, int N, int K)
{
  __shared__ _Float16 As[128][40];   // [m][k], pitch 40 halfs (TDM pad)
  __shared__ _Float16 Bs[128][40];   // [n][k], pitch 40 halfs (TDM pad)
  const int tid  = threadIdx.x;
  const int bm0  = blockIdx.x * 128;
  const int bn0  = blockIdx.y * 128;
  const int wave = tid >> 5;
  const int lane = tid & 31;
  const int lm   = lane & 15;
  const int hi   = lane >> 4;
  const int wm   = wave & 3;   // 4 waves over M
  const int wn   = wave >> 2;  // 2 waves over N
  const unsigned asl = (unsigned)(size_t)&As[0][0];
  const unsigned bsl = (unsigned)(size_t)&Bs[0][0];

  v8f acc[2][4];
#pragma unroll
  for (int i = 0; i < 2; ++i)
#pragma unroll
    for (int t = 0; t < 4; ++t)
#pragma unroll
      for (int r = 0; r < 8; ++r) acc[i][t][r] = 0.f;

  const int ar = tid >> 1;          // im2col A row within tile
  const int ac = (tid & 1) * 16;    // im2col A col chunk

  for (int k0 = 0; k0 < K; k0 += 32) {
    // ---- stage tiles into LDS via Tensor Data Mover ----
    if (wave == 0) {
      if (AMODE == AMODE_F16) {
        unsigned mrem = (unsigned)(M - bm0);
        tdm_load_2d_f16(asl, A16 + (size_t)bm0 * K + k0,
                        (unsigned)(K - k0), mrem, (unsigned)K, 32u, 128u);
      }
      tdm_load_2d_f16(bsl, B16t + (size_t)bn0 * K + k0,
                      (unsigned)(K - k0), (unsigned)(N - bn0), (unsigned)K,
                      32u, 128u);
      __builtin_amdgcn_s_wait_tensorcnt(0);
    }
    if (AMODE == AMODE_IMG) {   // im2col from NCHW fp32 image, patch row = (b,p)
      int gr = bm0 + ar;
      if (gr < M) {
        int bi = gr / 196, p = gr % 196;
        int py = p / 14, px = p % 14;
        const float* ib = Aimg + (size_t)bi * 3 * 224 * 224;
#pragma unroll
        for (int j = 0; j < 16; ++j) {
          int k = k0 + ac + j;
          int c = k >> 8, rem = k & 255;
          int dy = rem >> 4, dx = rem & 15;
          As[ar][ac + j] =
            (_Float16)ib[((size_t)c * 224 + py * 16 + dy) * 224 + px * 16 + dx];
        }
      } else {
#pragma unroll
        for (int j = 0; j < 16; ++j) As[ar][ac + j] = (_Float16)0.f;
      }
    }
    __syncthreads();

    // ---- fragments + WMMA ----
    v16h afr[2], bfr[4];
#pragma unroll
    for (int i = 0; i < 2; ++i) {
      int row = wm * 32 + i * 16 + lm;
      v8h lo = *(const v8h*)&As[row][hi * 8];
      v8h h8 = *(const v8h*)&As[row][16 + hi * 8];
      afr[i] = make_v16(lo, h8);
    }
#pragma unroll
    for (int t = 0; t < 4; ++t) {
      int col = wn * 64 + t * 16 + lm;
      v8h lo = *(const v8h*)&Bs[col][hi * 8];
      v8h h8 = *(const v8h*)&Bs[col][16 + hi * 8];
      bfr[t] = make_v16(lo, h8);
    }
#pragma unroll
    for (int i = 0; i < 2; ++i)
#pragma unroll
      for (int t = 0; t < 4; ++t)
        acc[i][t] = __builtin_amdgcn_wmma_f32_16x16x32_f16(
            false, afr[i], false, bfr[t], (short)0, acc[i][t], false, false);
    __syncthreads();
  }

  // ---- epilogue ----
#pragma unroll
  for (int i = 0; i < 2; ++i) {
#pragma unroll
    for (int t = 0; t < 4; ++t) {
      int gcol = bn0 + wn * 64 + t * 16 + lm;
#pragma unroll
      for (int r = 0; r < 8; ++r) {
        int grow = bm0 + wm * 32 + i * 16 + r + 8 * hi;
        if (grow >= M) continue;
        float v = acc[i][t][r];
        if (EMODE == EP_F16) {
          Ch[(size_t)grow * N + gcol] = (_Float16)v;
        } else if (EMODE == EP_GELU) {
          v += bias[gcol];
          float g = 0.5f * v * (1.0f + erff(v * 0.70710678118654752f));
          Ch[(size_t)grow * N + gcol] = (_Float16)g;
        } else if (EMODE == EP_RES) {
          Cf[(size_t)grow * N + gcol] += v + bias[gcol];
        } else { // EP_EMBED: patch token scatter + conv_b + pos + time
          int bi = grow / 196, p = grow % 196;
          float o = v + bias[gcol] + pos[(size_t)(1 + p) * DIMD + gcol] +
                    tem[(size_t)(bi % 20) * DIMD + gcol];
          Cf[((size_t)bi * NTOK + 1 + p) * DIMD + gcol] = o;
        }
      }
    }
  }
}

// ---------------------------------------------------------------------------
// Attention: one block per (batch, head). 4 waves, each owns 16-row q strips.
// qkv layout: [token][2304] f16 (q|k|v each 768 = 12 heads x 64)
// ---------------------------------------------------------------------------
__global__ __launch_bounds__(128)
void attn_kernel(const _Float16* __restrict__ qkv, _Float16* __restrict__ o16)
{
  __shared__ _Float16 P[4][16][224];   // per-wave probability strip (K pad 224)
  __shared__ _Float16 Vt[64][232];     // V transposed [d][key], key padded
  const int bh = blockIdx.x;
  const int b  = bh / NHEAD;
  const int h  = bh % NHEAD;
  const int tid  = threadIdx.x;
  const int wave = tid >> 5;
  const int lane = tid & 31;
  const int lm   = lane & 15;
  const int hi   = lane >> 4;
  const int base = b * NTOK;
  const _Float16* qp = qkv + (size_t)h * DHEAD;
  const _Float16* kp = qkv + 768  + (size_t)h * DHEAD;
  const _Float16* vp = qkv + 1536 + (size_t)h * DHEAD;

  // cooperative transpose of V into LDS (rows >= 199 zero-filled)
  for (int c = tid; c < 224 * 8; c += 128) {
    int row = c >> 3, ch = (c & 7) * 8;
    if (row < NTOK) {
      v8h val = *(const v8h*)(vp + (size_t)(base + row) * 2304 + ch);
#pragma unroll
      for (int j = 0; j < 8; ++j) Vt[ch + j][row] = val[j];
    } else {
#pragma unroll
      for (int j = 0; j < 8; ++j) Vt[ch + j][row] = (_Float16)0.f;
    }
  }
  __syncthreads();

  for (int rt = wave; rt < 13; rt += 4) {
    const int m0 = rt * 16;
    // q A-fragments (K = 64 -> 2 k-steps)
    int qrow = m0 + lm; if (qrow > NTOK - 1) qrow = NTOK - 1;
    const _Float16* qr = qp + (size_t)(base + qrow) * 2304;
    v16h qa[2];
#pragma unroll
    for (int s = 0; s < 2; ++s) {
      v8h lo = *(const v8h*)(qr + s * 32 + hi * 8);
      v8h h8 = *(const v8h*)(qr + s * 32 + 16 + hi * 8);
      qa[s] = make_v16(lo, h8);
    }
    // scores: 13 key tiles of 16
    v8f sc[13];
#pragma unroll
    for (int kt = 0; kt < 13; ++kt)
#pragma unroll
      for (int r = 0; r < 8; ++r) sc[kt][r] = 0.f;
#pragma unroll
    for (int kt = 0; kt < 13; ++kt) {
      int key = kt * 16 + lm;
      bool kvld = key < NTOK;
      int keyc = kvld ? key : NTOK - 1;
      const _Float16* kr = kp + (size_t)(base + keyc) * 2304;
#pragma unroll
      for (int s = 0; s < 2; ++s) {
        v8h lo = *(const v8h*)(kr + s * 32 + hi * 8);
        v8h h8 = *(const v8h*)(kr + s * 32 + 16 + hi * 8);
        v16h kb;
#pragma unroll
        for (int j = 0; j < 8; ++j) {
          kb[j]     = kvld ? lo[j] : (_Float16)0.f;
          kb[j + 8] = kvld ? h8[j] : (_Float16)0.f;
        }
        sc[kt] = __builtin_amdgcn_wmma_f32_16x16x32_f16(
            false, qa[s], false, kb, (short)0, sc[kt], false, false);
      }
    }
    // exact softmax per row; normalized probs written to LDS (ref order)
#pragma unroll
    for (int r = 0; r < 8; ++r) {
      float mx = -3.0e38f;
#pragma unroll
      for (int kt = 0; kt < 13; ++kt) {
        float v = sc[kt][r] * 0.125f;           // DH^-0.5
        int col = kt * 16 + lm;
        if (col >= NTOK) v = -1.0e30f;
        sc[kt][r] = v;
        mx = fmaxf(mx, v);
      }
#pragma unroll
      for (int off = 8; off; off >>= 1) mx = fmaxf(mx, __shfl_xor(mx, off, 32));
      float sum = 0.f;
#pragma unroll
      for (int kt = 0; kt < 13; ++kt) {
        float e = __expf(sc[kt][r] - mx);
        sc[kt][r] = e;
        sum += e;
      }
#pragma unroll
      for (int off = 8; off; off >>= 1) sum += __shfl_xor(sum, off, 32);
      float rinv = 1.0f / sum;
      int m = r + 8 * hi;
#pragma unroll
      for (int kt = 0; kt < 13; ++kt)
        P[wave][m][kt * 16 + lm] = (_Float16)(sc[kt][r] * rinv);
      P[wave][m][208 + lm] = (_Float16)0.f;     // zero K-pad 208..223
    }
    // cross-lane LDS RAW inside a wave: wait LDS writes
    asm volatile("s_wait_dscnt 0x0" ::: "memory");

    // O = P @ V   (K = 224, 7 k-steps)
    v8f oa[4];
#pragma unroll
    for (int t = 0; t < 4; ++t)
#pragma unroll
      for (int r = 0; r < 8; ++r) oa[t][r] = 0.f;
#pragma unroll
    for (int s = 0; s < 7; ++s) {
      const _Float16* pr = &P[wave][lm][s * 32];
      v8h lo = *(const v8h*)(pr + hi * 8);
      v8h h8 = *(const v8h*)(pr + 16 + hi * 8);
      v16h pa = make_v16(lo, h8);
#pragma unroll
      for (int t = 0; t < 4; ++t) {
        int d = t * 16 + lm;
        const _Float16* vr = &Vt[d][s * 32 + hi * 8];
        v8h vlo = *(const v8h*)vr;
        v8h vh8 = *(const v8h*)(vr + 16);
        v16h vb = make_v16(vlo, vh8);
        oa[t] = __builtin_amdgcn_wmma_f32_16x16x32_f16(
            false, pa, false, vb, (short)0, oa[t], false, false);
      }
    }
#pragma unroll
    for (int t = 0; t < 4; ++t)
#pragma unroll
      for (int r = 0; r < 8; ++r) {
        int m = r + 8 * hi;
        int tok = m0 + m;
        if (tok < NTOK)
          o16[(size_t)(base + tok) * DIMD + h * DHEAD + t * 16 + lm] =
              (_Float16)oa[t][r];
      }
  }
}

// ---------------------------------------------------------------------------
// LayerNorm (fp32 in -> f16 out), one 768-row per block of 256 threads
// ---------------------------------------------------------------------------
__global__ __launch_bounds__(256)
void ln_kernel(const float* __restrict__ x, const float* __restrict__ gam,
               const float* __restrict__ bet, _Float16* __restrict__ out)
{
  const int row = blockIdx.x;
  const int tid = threadIdx.x;
  const float* xr = x + (size_t)row * DIMD;
  float v0 = xr[tid], v1 = xr[tid + 256], v2 = xr[tid + 512];
  __shared__ float sred[8];
  __shared__ float sbc;
  float s = v0 + v1 + v2;
#pragma unroll
  for (int off = 16; off; off >>= 1) s += __shfl_xor(s, off, 32);
  if ((tid & 31) == 0) sred[tid >> 5] = s;
  __syncthreads();
  if (tid == 0) {
    float t = 0.f;
    for (int i = 0; i < 8; ++i) t += sred[i];
    sbc = t * (1.0f / 768.0f);
  }
  __syncthreads();
  const float mean = sbc;
  float d0 = v0 - mean, d1 = v1 - mean, d2 = v2 - mean;
  float q = d0 * d0 + d1 * d1 + d2 * d2;
#pragma unroll
  for (int off = 16; off; off >>= 1) q += __shfl_xor(q, off, 32);
  __syncthreads();
  if ((tid & 31) == 0) sred[tid >> 5] = q;
  __syncthreads();
  if (tid == 0) {
    float t = 0.f;
    for (int i = 0; i < 8; ++i) t += sred[i];
    sbc = rsqrtf(t * (1.0f / 768.0f) + 1e-5f);
  }
  __syncthreads();
  const float rstd = sbc;
  out[(size_t)row * DIMD + tid]       = (_Float16)(d0 * rstd * gam[tid]       + bet[tid]);
  out[(size_t)row * DIMD + tid + 256] = (_Float16)(d1 * rstd * gam[tid + 256] + bet[tid + 256]);
  out[(size_t)row * DIMD + tid + 512] = (_Float16)(d2 * rstd * gam[tid + 512] + bet[tid + 512]);
}

// weight fp32 (KxN) -> f16 N-major (Wt[n*K + k])
__global__ __launch_bounds__(256)
void convert_w_t(const float* __restrict__ W, _Float16* __restrict__ Wt,
                 int K, int N)
{
  int k = blockIdx.x * 256 + threadIdx.x;
  int n = blockIdx.y;
  Wt[(size_t)n * K + k] = (_Float16)W[(size_t)k * N + n];
}

// conv_w OIHW fp32 is already N-major (n = out-ch, k = c*256+dy*16+dx): copy/convert
__global__ __launch_bounds__(256)
void convert_w_c(const float* __restrict__ W, _Float16* __restrict__ Wt, int total)
{
  int i = blockIdx.x * 256 + threadIdx.x;
  if (i < total) Wt[i] = (_Float16)W[i];
}

// cls token (pos+time) and face tokens (+time) into x
__global__ __launch_bounds__(256)
void fill_special(const float* __restrict__ face, const float* __restrict__ pos,
                  const float* __restrict__ tem, const float* __restrict__ cls,
                  float* __restrict__ x)
{
  const int b = blockIdx.x;           // 0..39
  const int t = b % 20;
  for (int d = threadIdx.x; d < DIMD; d += 256) {
    float te = tem[(size_t)t * DIMD + d];
    x[((size_t)b * NTOK + 0)   * DIMD + d] = cls[d] + pos[d] + te;
    x[((size_t)b * NTOK + 197) * DIMD + d] = face[((size_t)b * 2 + 0) * DIMD + d] + te;
    x[((size_t)b * NTOK + 198) * DIMD + d] = face[((size_t)b * 2 + 1) * DIMD + d] + te;
  }
}

// tuple output: x[:,0], x[:,197], x[:,198], x[:,1:197]
__global__ __launch_bounds__(256)
void gather_out(const float* __restrict__ x, float* __restrict__ out)
{
  const int row = blockIdx.x;          // 0..7959
  const int b = row / NTOK, j = row % NTOK;
  size_t dst;
  if (j == 0)        dst = (size_t)b * DIMD;
  else if (j == 197) dst = (size_t)(40 + b) * DIMD;
  else if (j == 198) dst = (size_t)(80 + b) * DIMD;
  else               dst = (size_t)(120 + b * 196 + (j - 1)) * DIMD;
  for (int d = threadIdx.x; d < DIMD; d += 256)
    out[dst + d] = x[(size_t)row * DIMD + d];
}

// ---------------------------------------------------------------------------
extern "C" void kernel_launch(void* const* d_in, const int* in_sizes, int n_in,
                              void* d_out, int out_size, void* d_ws, size_t ws_size,
                              hipStream_t stream)
{
  (void)in_sizes; (void)n_in; (void)out_size; (void)ws_size;
  const float* img      = (const float*)d_in[0];
  const float* face     = (const float*)d_in[1];
  const float* conv_w   = (const float*)d_in[2];
  const float* conv_b   = (const float*)d_in[3];
  const float* pos_emb  = (const float*)d_in[4];
  const float* time_emb = (const float*)d_in[5];
  const float* cls_tok  = (const float*)d_in[6];
  const float* ln1_s    = (const float*)d_in[7];
  const float* ln1_b    = (const float*)d_in[8];
  const float* qkv_w    = (const float*)d_in[9];
  const float* out_w    = (const float*)d_in[10];
  const float* out_b    = (const float*)d_in[11];
  const float* ln2_s    = (const float*)d_in[12];
  const float* ln2_b    = (const float*)d_in[13];
  const float* ff_w1    = (const float*)d_in[14];
  const float* ff_b1    = (const float*)d_in[15];
  const float* ff_w2    = (const float*)d_in[16];
  const float* ff_b2    = (const float*)d_in[17];

  // workspace layout (sizes 256B-aligned; o16 last with slack so TDM edge-tile
  // over-reads of A stay inside d_ws)
  char* ws = (char*)d_ws;
  const size_t XB = (size_t)TROWS * DIMD * 4;   // x        24,453,120
  const size_t HB = (size_t)TROWS * DIMD * 2;   // h16      12,226,560
  const size_t BB = (size_t)TROWS * MLPD * 2;   // qkv/mid  48,906,240
  const size_t WB = (size_t)DIMD * MLPD * 2;    // wbuf      4,718,592
  float*    x     = (float*)ws;
  _Float16* h16   = (_Float16*)(ws + XB);
  _Float16* big16 = (_Float16*)(ws + XB + HB);
  _Float16* wbuf  = (_Float16*)(ws + XB + HB + BB);
  _Float16* o16   = (_Float16*)(ws + XB + HB + BB + WB);

  // patch embed (im2col GEMM; conv_w converted to f16, already N-major)
  convert_w_c<<<2304, 256, 0, stream>>>(conv_w, wbuf, DIMD * DIMD);
  gemm_wmma<AMODE_IMG, EP_EMBED><<<dim3(62, 6), 256, 0, stream>>>(
      nullptr, img, wbuf, conv_b, x, nullptr, pos_emb, time_emb,
      7840, DIMD, DIMD);
  fill_special<<<40, 256, 0, stream>>>(face, pos_emb, time_emb, cls_tok, x);

  for (int i = 0; i < 12; ++i) {
    ln_kernel<<<TROWS, 256, 0, stream>>>(x, ln1_s + (size_t)i * DIMD,
                                         ln1_b + (size_t)i * DIMD, h16);
    convert_w_t<<<dim3(3, 2304), 256, 0, stream>>>(
        qkv_w + (size_t)i * DIMD * 2304, wbuf, DIMD, 2304);
    gemm_wmma<AMODE_F16, EP_F16><<<dim3(63, 18), 256, 0, stream>>>(
        h16, nullptr, wbuf, nullptr, nullptr, big16, nullptr, nullptr,
        TROWS, 2304, DIMD);
    attn_kernel<<<BTN * NHEAD, 128, 0, stream>>>(big16, o16);
    convert_w_t<<<dim3(3, DIMD), 256, 0, stream>>>(
        out_w + (size_t)i * DIMD * DIMD, wbuf, DIMD, DIMD);
    gemm_wmma<AMODE_F16, EP_RES><<<dim3(63, 6), 256, 0, stream>>>(
        o16, nullptr, wbuf, out_b + (size_t)i * DIMD,
        x, nullptr, nullptr, nullptr, TROWS, DIMD, DIMD);
    ln_kernel<<<TROWS, 256, 0, stream>>>(x, ln2_s + (size_t)i * DIMD,
                                         ln2_b + (size_t)i * DIMD, h16);
    convert_w_t<<<dim3(3, MLPD), 256, 0, stream>>>(
        ff_w1 + (size_t)i * DIMD * MLPD, wbuf, DIMD, MLPD);
    gemm_wmma<AMODE_F16, EP_GELU><<<dim3(63, 24), 256, 0, stream>>>(
        h16, nullptr, wbuf, ff_b1 + (size_t)i * MLPD,
        nullptr, big16, nullptr, nullptr, TROWS, MLPD, DIMD);
    convert_w_t<<<dim3(12, DIMD), 256, 0, stream>>>(
        ff_w2 + (size_t)i * MLPD * DIMD, wbuf, MLPD, DIMD);
    gemm_wmma<AMODE_F16, EP_RES><<<dim3(63, 6), 256, 0, stream>>>(
        big16, nullptr, wbuf, ff_b2 + (size_t)i * DIMD,
        x, nullptr, nullptr, nullptr, TROWS, DIMD, MLPD);
  }
  gather_out<<<TROWS, 256, 0, stream>>>(x, (float*)d_out);
}
